// RandomMaskSubgraphs_74079595921840
// MI455X (gfx1250) — compile-verified, matrix operand order checked by reading, not run on hardware
//
#include <hip/hip_runtime.h>
#include <hip/hip_bf16.h>
#include <stdint.h>

// ---------------------------------------------------------------------------
// RandomMaskSubgraphs pipeline for MI455X (gfx1250, wave32).
//
// CDNA5 features used:
//  * Tensor Data Mover (tensor_load_to_lds) double-buffered streaming of the
//    edge lists into LDS, synced with s_wait_tensorcnt   (edge mask passes)
//  * v_wmma_f32_16x16x32_f16 "B=ones" tensor-core reductions (masked-node
//    count, unique-flag counts), fed by contiguous b128 loads (no predication)
//  * wave32 shuffle scans; single-wave workgroups where TDM/WMMA wants a full
//    wave with EXEC all-ones
//  * bitonic sort with fused LDS stages: 134 MB key array is resident in the
//    192 MB global L2; j<=2048 passes run entirely in LDS (32 KB tiles)
// ---------------------------------------------------------------------------

#define NTOT   150000       // N_USERS + N_ITEMS
#define NTOT_PAD 151552     // 37 * 4096 (zero-padded masked array)
#define EEDG   4000000      // E
#define NSEEDS 500
#define SAMPN  15000        // int(N * 0.1)
#define HLEN   (3 * (long)EEDG + NTOT)   // 12,150,000 hash entries
#define SORTN  (1u << 24)                // padded sort size (pad value = 0,
                                         // 0 already occurs -> unique set unchanged)
#define CHUNK  1024                      // edges per TDM chunk

typedef _Float16 v16h __attribute__((ext_vector_type(16)));
typedef float    v8f  __attribute__((ext_vector_type(8)));
typedef unsigned int u32x4 __attribute__((ext_vector_type(4)));
typedef int          i32x4 __attribute__((ext_vector_type(4)));
typedef int          i32x8 __attribute__((ext_vector_type(8)));
typedef unsigned long long u64x2 __attribute__((ext_vector_type(2)));

// ------------------------------- RNG ---------------------------------------
__device__ __forceinline__ unsigned long long splitmix64(unsigned long long x) {
  x += 0x9E3779B97F4A7C15ull;
  x = (x ^ (x >> 30)) * 0xBF58476D1CE4E5B9ull;
  x = (x ^ (x >> 27)) * 0x94D049BB133111EBull;
  return x ^ (x >> 31);
}

__device__ __forceinline__ long long rand_idx(unsigned long long seed, long e,
                                              long long tem) {
  unsigned long long x = splitmix64(seed ^ (unsigned long long)e);
  double u = (double)(x >> 11) * (1.0 / 9007199254740992.0);  // [0,1)
  long long i = (long long)(u * (double)tem);
  if (i > tem - 1) i = tem - 1;
  if (i < 0) i = 0;
  return i;
}

// ------------------------- TDM 1-D chunk load ------------------------------
// D# per CDNA5 ISA ch.8: group0 = {count=1, lds_addr, global_addr, type=2},
// group1 = {data_size=4B, tensor_dim0=elems, tensor_dim1=1, tile_dim0=elems,
//           tile_dim1=1, stride0=elems}. Groups 2/3 zero (<=2D tensor).
__device__ __forceinline__ void tdm_load_i32(uint32_t lds_byte_addr,
                                             const int* gsrc, uint32_t elems) {
  unsigned long long ga = (unsigned long long)(const void*)gsrc;
  u32x4 g0;
  g0[0] = 1u;                                   // count=1, is_restore=0
  g0[1] = lds_byte_addr;                        // lds_addr
  g0[2] = (uint32_t)ga;                         // global_addr lo
  g0[3] = (uint32_t)(ga >> 32) | (2u << 30);    // global_addr hi | type=2
  i32x8 g1;
  g1[0] = (int)(2u << 16);                      // data_size = 2 -> 4 bytes
  g1[1] = (int)((elems & 0xFFFFu) << 16);       // tensor_dim0[15:0]
  g1[2] = (int)(((elems >> 16) & 0xFFFFu) | (1u << 16)); // tdim0 hi | tdim1=1 lo
  g1[3] = (int)((elems & 0xFFFFu) << 16);       // tdim1 hi=0 | tile_dim0=elems
  g1[4] = 1;                                    // tile_dim1=1, tile_dim2=0
  g1[5] = (int)elems;                           // tensor_dim0_stride lo
  g1[6] = 0;                                    // stride0 hi | stride1 lo
  g1[7] = 0;                                    // stride1 hi
  i32x4 gz = {0, 0, 0, 0};
#if defined(__clang_major__) && (__clang_major__ >= 23)
  i32x8 gz8 = {0, 0, 0, 0, 0, 0, 0, 0};
  __builtin_amdgcn_tensor_load_to_lds(g0, g1, gz, gz, gz8, 0);
#else
  __builtin_amdgcn_tensor_load_to_lds(g0, g1, gz, gz, 0);
#endif
}

// ------------------- WMMA tensor-core count reduction ----------------------
// acc = A(16x32 flags) x B(ones) + acc ; total(D) = 16 * sum(flags).
__device__ __forceinline__ float wave_wmma_total(v8f acc) {
  float s = 0.f;
#pragma unroll
  for (int r = 0; r < 8; ++r) s += acc[r];
  for (int off = 16; off > 0; off >>= 1) s += __shfl_xor(s, off, 32);
  return s * (1.0f / 16.0f);
}

// ------------------------------ kernels ------------------------------------

__global__ void k_init_nodes(uint32_t* frontier, uint32_t* hits,
                             uint32_t* masked, float* deg) {
  int i = blockIdx.x * blockDim.x + threadIdx.x;
  if (i < NTOT_PAD) {
    masked[i] = 0u;                       // padded region must be zero
    if (i < NTOT) { frontier[i] = 0u; hits[i] = 0u; deg[i] = 0.f; }
  }
}

__global__ void k_init_alive(uint8_t* alive) {
  long e = (long)blockIdx.x * blockDim.x + threadIdx.x;
  if (e < EEDG) alive[e] = 1u;
}

__global__ void k_seed(const int* __restrict__ seeds, uint32_t* frontier,
                       uint32_t* masked) {
  int t = blockIdx.x * blockDim.x + threadIdx.x;
  if (t < NSEEDS) {
    int s = seeds[t];
    frontier[s] = 1u;
    masked[s] = 1u;
  }
}

// Edge mask pass: TDM double-buffered streaming of rows/cols into LDS.
__global__ __launch_bounds__(32)
void k_edge_pass(const int* __restrict__ rows, const int* __restrict__ cols,
                 const uint32_t* __restrict__ frontier,
                 uint32_t* __restrict__ hits, uint8_t* __restrict__ alive,
                 int collect) {
  __shared__ int rbuf[2][CHUNK];
  __shared__ int cbuf[2][CHUNK];
  const int nchunks = (EEDG + CHUNK - 1) / CHUNK;
  const int lane = (int)threadIdx.x;
  const int stride = (int)gridDim.x;
  int c = (int)blockIdx.x;
  if (c >= nchunks) return;

  uint32_t lr[2], lc[2];
  lr[0] = (uint32_t)(unsigned long long)(void*)&rbuf[0][0];
  lr[1] = (uint32_t)(unsigned long long)(void*)&rbuf[1][0];
  lc[0] = (uint32_t)(unsigned long long)(void*)&cbuf[0][0];
  lc[1] = (uint32_t)(unsigned long long)(void*)&cbuf[1][0];

  { // prologue: chunk c into buffer 0
    uint32_t s = (uint32_t)c * CHUNK;
    uint32_t e = (uint32_t)EEDG - s; if (e > CHUNK) e = CHUNK;
    tdm_load_i32(lr[0], rows + s, e);
    tdm_load_i32(lc[0], cols + s, e);
  }
  int buf = 0;
  for (; c < nchunks; c += stride) {
    int nc = c + stride;
    uint32_t s = (uint32_t)c * CHUNK;
    uint32_t elems = (uint32_t)EEDG - s; if (elems > CHUNK) elems = CHUNK;
    if (nc < nchunks) {
      uint32_t s2 = (uint32_t)nc * CHUNK;
      uint32_t e2 = (uint32_t)EEDG - s2; if (e2 > CHUNK) e2 = CHUNK;
      tdm_load_i32(lr[buf ^ 1], rows + s2, e2);
      tdm_load_i32(lc[buf ^ 1], cols + s2, e2);
      __builtin_amdgcn_s_wait_tensorcnt(2);  // oldest 2 (current chunk) done
    } else {
      __builtin_amdgcn_s_wait_tensorcnt(0);
    }
    for (uint32_t i = (uint32_t)lane; i < elems; i += 32u) {
      int r = rbuf[buf][i];
      int cc = cbuf[buf][i];
      long e64 = (long)s + (long)i;
      if (alive[e64]) {
        if ((frontier[r] | frontier[cc]) != 0u) {
          alive[e64] = 0u;
          if (collect) { atomicOr(&hits[r], 1u); atomicOr(&hits[cc], 1u); }
        }
      }
    }
    buf ^= 1;
  }
}

__global__ void k_frontier_update(uint32_t* frontier, const uint32_t* hits,
                                  uint32_t* masked) {
  int i = blockIdx.x * blockDim.x + threadIdx.x;
  if (i < NTOT) {
    uint32_t f = (hits[i] != 0u) ? 1u : 0u;
    frontier[i] = f;
    if (f) masked[i] = 1u;
  }
}

__global__ void k_sample(uint32_t* masked) {
  int t = blockIdx.x * blockDim.x + threadIdx.x;
  if (t < SAMPN) {
    unsigned long long x = splitmix64(0xA5A5F00DB16B00B5ull ^ (unsigned long long)t);
    uint32_t node = (uint32_t)(x % (unsigned long long)NTOT);
    atomicOr(&masked[node], 1u);
  }
}

// WMMA count of masked flags; masked is zero-padded to NTOT_PAD so there are
// no bounds checks: 4 x b128 loads per lane, then cvt -> v_wmma.
__global__ __launch_bounds__(32)
void k_mask_count(const uint32_t* __restrict__ masked, uint32_t* blkcnt) {
  int lane = (int)threadIdx.x;
  v8f acc = {};
  v16h ones;
#pragma unroll
  for (int i = 0; i < 16; ++i) ones[i] = (_Float16)1.0f;
  for (int t = 0; t < 8; ++t) {
    const uint4* q =
        (const uint4*)(masked + (long)blockIdx.x * 4096 + t * 512 + lane * 16);
    uint4 a0 = q[0], a1 = q[1], a2 = q[2], a3 = q[3];
    v16h a;
    a[0]  = (_Float16)(float)a0.x; a[1]  = (_Float16)(float)a0.y;
    a[2]  = (_Float16)(float)a0.z; a[3]  = (_Float16)(float)a0.w;
    a[4]  = (_Float16)(float)a1.x; a[5]  = (_Float16)(float)a1.y;
    a[6]  = (_Float16)(float)a1.z; a[7]  = (_Float16)(float)a1.w;
    a[8]  = (_Float16)(float)a2.x; a[9]  = (_Float16)(float)a2.y;
    a[10] = (_Float16)(float)a2.z; a[11] = (_Float16)(float)a2.w;
    a[12] = (_Float16)(float)a3.x; a[13] = (_Float16)(float)a3.y;
    a[14] = (_Float16)(float)a3.z; a[15] = (_Float16)(float)a3.w;
    acc = __builtin_amdgcn_wmma_f32_16x16x32_f16(false, a, false, ones,
                                                 (short)0, acc, false, false);
  }
  float tot = wave_wmma_total(acc);
  if (lane == 0) blkcnt[blockIdx.x] = (uint32_t)(tot + 0.5f);
}

// wave32 shuffle scan (exclusive, in place), 32 elements per step + carry.
__global__ __launch_bounds__(32)
void k_scan_wave(uint32_t* cnt, int n, uint32_t* total) {
  int lane = (int)threadIdx.x;
  uint32_t carry = 0;
  for (int base = 0; base < n; base += 32) {
    int idx = base + lane;
    uint32_t v = (idx < n) ? cnt[idx] : 0u;
    uint32_t x = v;
    for (int d = 1; d < 32; d <<= 1) {
      uint32_t o = __shfl_up(x, d, 32);
      if (lane >= d) x += o;
    }
    if (idx < n) cnt[idx] = carry + (x - v);
    carry += __shfl(x, 31, 32);
  }
  if (lane == 0 && total) *total = carry;
}

// Stable partition: masked node ids first (ascending), then unmasked.
__global__ __launch_bounds__(32)
void k_order_write(const uint32_t* __restrict__ masked,
                   const uint32_t* __restrict__ blkoff,
                   const uint32_t* __restrict__ scal, int* order, int n) {
  int lane = (int)threadIdx.x;
  long base = (long)blockIdx.x * 4096 + (long)lane * 128;
  uint32_t tem = scal[0];
  uint32_t ct = 0;
  for (int i = 0; i < 128; ++i) ct += masked[base + i];   // padded: no guard
  uint32_t v = ct;                                        // wave32 scan
  for (int d = 1; d < 32; d <<= 1) {
    uint32_t o = __shfl_up(v, d, 32);
    if (lane >= d) v += o;
  }
  uint32_t mrank = blkoff[blockIdx.x] + (v - ct);
  for (int i = 0; i < 128; ++i) {
    long idx = base + i;
    if (idx >= n) break;
    if (masked[idx] != 0u) { order[mrank] = (int)idx; mrank++; }
    else                   { order[tem + (uint32_t)idx - mrank] = (int)idx; }
  }
}

__global__ void k_deg(const int* __restrict__ rows,
                      const uint8_t* __restrict__ alive, float* deg) {
  long e = (long)blockIdx.x * blockDim.x + threadIdx.x;
  if (e < EEDG && alive[e]) atomicAdd(&deg[rows[e]], 1.0f);
}

__global__ void k_dnorm(const float* deg, float* dnorm) {
  int i = blockIdx.x * blockDim.x + threadIdx.x;
  if (i < NTOT) dnorm[i] = rsqrtf(deg[i] + 1e-12f);
}

__global__ void k_enc_out(const int* __restrict__ rows,
                          const int* __restrict__ cols,
                          const uint8_t* __restrict__ alive,
                          const float* __restrict__ dnorm, float* out) {
  long e = ((long)blockIdx.x * blockDim.x + threadIdx.x) * 4;
  if (e < EEDG) {
    int4 r4 = *(const int4*)(rows + e);
    int4 c4 = *(const int4*)(cols + e);
    uchar4 a4 = *(const uchar4*)(alive + e);
    float4 fr = {(float)r4.x, (float)r4.y, (float)r4.z, (float)r4.w};
    float4 fc = {(float)c4.x, (float)c4.y, (float)c4.z, (float)c4.w};
    float4 ev;
    ev.x = a4.x ? dnorm[r4.x] * dnorm[c4.x] : 0.0f;
    ev.y = a4.y ? dnorm[r4.y] * dnorm[c4.y] : 0.0f;
    ev.z = a4.z ? dnorm[r4.z] * dnorm[c4.z] : 0.0f;
    ev.w = a4.w ? dnorm[r4.w] * dnorm[c4.w] : 0.0f;
    *(float4*)(out + e)              = fr;
    *(float4*)(out + (long)EEDG + e) = fc;
    *(float4*)(out + 2L * EEDG + e)  = ev;
  }
}

__global__ void k_dec_zero(float* out) {
  long i = ((long)blockIdx.x * blockDim.x + threadIdx.x) * 4;
  if (i < 3L * HLEN) {
    float4 z = {0.f, 0.f, 0.f, 0.f};
    *(float4*)(out + 3L * EEDG + i) = z;
  }
}

__global__ void k_hashgen(const int* __restrict__ rows,
                          const int* __restrict__ cols,
                          const uint8_t* __restrict__ alive,
                          const int* __restrict__ order,
                          const uint32_t* __restrict__ scal,
                          unsigned long long* keys) {
  long j = (long)blockIdx.x * blockDim.x + threadIdx.x;
  if (j >= (long)SORTN) return;
  long long tem = (long long)scal[0];
  if (tem < 1) tem = 1;
  unsigned long long h;
  if (j < (long)EEDG) {
    long long tr = order[rand_idx(0x1ED0C0FFEEull, j, tem)];
    long long tc = order[rand_idx(0xC0DEC0DE42ull, j, tem)];
    h = (unsigned long long)(tr * (long long)NTOT + tc);
  } else if (j < 2L * EEDG) {
    long e = j - EEDG;
    long long tr = order[rand_idx(0x1ED0C0FFEEull, e, tem)];
    long long tc = order[rand_idx(0xC0DEC0DE42ull, e, tem)];
    h = (unsigned long long)(tc * (long long)NTOT + tr);
  } else if (j < 2L * EEDG + NTOT) {
    long a = j - 2L * EEDG;
    h = (unsigned long long)a * (unsigned long long)(NTOT + 1);
  } else if (j < HLEN) {
    long e = j - 2L * EEDG - NTOT;
    h = alive[e] ? (unsigned long long)((long long)rows[e] * NTOT + cols[e]) : 0ull;
  } else {
    h = 0ull;  // pad: duplicates of existing hash 0 -> unique set unchanged
  }
  keys[j] = h;
}

// global bitonic compare-exchange pass (only for j >= 4096)
__global__ void k_bitonic(unsigned long long* a, unsigned j, unsigned k) {
  unsigned i = blockIdx.x * blockDim.x + threadIdx.x;
  unsigned ixj = i ^ j;
  if (ixj > i) {
    bool up = ((i & k) == 0u);
    unsigned long long x = a[i], y = a[ixj];
    if ((x > y) == up) { a[i] = y; a[ixj] = x; }
  }
}

// fused LDS bitonic: all stages k=2..4096 on a 4096-key tile (78 passes)
__global__ __launch_bounds__(512)
void k_bitonic_presort(unsigned long long* a) {
  __shared__ unsigned long long s[4096];
  unsigned base = blockIdx.x * 4096u;
  for (int i = threadIdx.x; i < 4096; i += 512) s[i] = a[base + i];
  __syncthreads();
  for (unsigned k = 2; k <= 4096; k <<= 1) {
    for (unsigned j = k >> 1; j > 0; j >>= 1) {
      for (unsigned t = threadIdx.x; t < 2048; t += 512) {
        unsigned i = ((t & ~(j - 1u)) << 1) | (t & (j - 1u));
        unsigned ixj = i | j;
        bool up = (((base + i) & k) == 0u);
        unsigned long long x = s[i], y = s[ixj];
        if ((x > y) == up) { s[i] = y; s[ixj] = x; }
      }
      __syncthreads();
    }
  }
  for (int i = threadIdx.x; i < 4096; i += 512) a[base + i] = s[i];
}

// fused LDS bitonic finisher for a given k: passes j=2048..1 on a 4096 tile
__global__ __launch_bounds__(512)
void k_bitonic_lds(unsigned long long* a, unsigned k) {
  __shared__ unsigned long long s[4096];
  unsigned base = blockIdx.x * 4096u;
  for (int i = threadIdx.x; i < 4096; i += 512) s[i] = a[base + i];
  __syncthreads();
  for (unsigned j = 2048; j > 0; j >>= 1) {
    for (unsigned t = threadIdx.x; t < 2048; t += 512) {
      unsigned i = ((t & ~(j - 1u)) << 1) | (t & (j - 1u));
      unsigned ixj = i | j;
      bool up = (((base + i) & k) == 0u);
      unsigned long long x = s[i], y = s[ixj];
      if ((x > y) == up) { s[i] = y; s[ixj] = x; }
    }
    __syncthreads();
  }
  for (int i = threadIdx.x; i < 4096; i += 512) a[base + i] = s[i];
}

// WMMA count of "first occurrence" flags over sorted keys (4096/block),
// fed by 16B vector loads; previous key carried in a register.
__global__ __launch_bounds__(32)
void k_uniq_count(const unsigned long long* __restrict__ keys, uint32_t* blkcnt) {
  int lane = (int)threadIdx.x;
  v8f acc = {};
  v16h ones;
#pragma unroll
  for (int i = 0; i < 16; ++i) ones[i] = (_Float16)1.0f;
  for (int t = 0; t < 8; ++t) {
    long b0 = (long)blockIdx.x * 4096 + (long)t * 512 + (long)lane * 16;
    const u64x2* q = (const u64x2*)(keys + b0);
    unsigned long long kv[16];
#pragma unroll
    for (int i = 0; i < 8; ++i) {
      u64x2 p = q[i];
      kv[2 * i] = p[0];
      kv[2 * i + 1] = p[1];
    }
    unsigned long long prev = (b0 == 0) ? (kv[0] ^ 1ull) : keys[b0 - 1];
    v16h a;
#pragma unroll
    for (int i = 0; i < 16; ++i) {
      a[i] = (_Float16)((kv[i] != prev) ? 1.0f : 0.0f);
      prev = kv[i];
    }
    acc = __builtin_amdgcn_wmma_f32_16x16x32_f16(false, a, false, ones,
                                                 (short)0, acc, false, false);
  }
  float tot = wave_wmma_total(acc);
  if (lane == 0) blkcnt[blockIdx.x] = (uint32_t)(tot + 0.5f);
}

__global__ __launch_bounds__(32)
void k_uniq_write(const unsigned long long* __restrict__ keys,
                  const uint32_t* __restrict__ blkoff, float* out) {
  int lane = (int)threadIdx.x;
  long base = (long)blockIdx.x * 4096 + (long)lane * 128;
  unsigned long long first = keys[base];
  unsigned long long prev0 = (base == 0) ? (first ^ 1ull) : keys[base - 1];

  uint32_t ct = 0;
  {
    unsigned long long prev = prev0;
    for (int i = 0; i < 128; i += 2) {
      u64x2 p = *(const u64x2*)(keys + base + i);
      ct += (p[0] != prev);
      ct += (p[1] != p[0]);
      prev = p[1];
    }
  }
  uint32_t w = ct;
  for (int d = 1; d < 32; d <<= 1) {
    uint32_t o = __shfl_up(w, d, 32);
    if (lane >= d) w += o;
  }
  uint32_t rank = blkoff[blockIdx.x] + (w - ct);
  float* dr = out + 3L * EEDG;
  float* dc = dr + HLEN;
  float* dv = dc + HLEN;
  unsigned long long prev = prev0;
  for (int i = 0; i < 128; i += 2) {
    u64x2 p = *(const u64x2*)(keys + base + i);
#pragma unroll
    for (int h = 0; h < 2; ++h) {
      unsigned long long v = p[h];
      if (v != prev) {
        if ((long)rank < HLEN) {
          dr[rank] = (float)(v / (unsigned long long)NTOT);
          dc[rank] = (float)(v % (unsigned long long)NTOT);
          dv[rank] = 1.0f;
        }
        rank++;
      }
      prev = v;
    }
  }
}

// ------------------------------- host --------------------------------------
extern "C" void kernel_launch(void* const* d_in, const int* in_sizes, int n_in,
                              void* d_out, int out_size, void* d_ws, size_t ws_size,
                              hipStream_t stream) {
  (void)in_sizes; (void)n_in; (void)out_size; (void)ws_size;
  // adj_vals (d_in[0]) is unused by the reference computation (enc values are 1).
  const int* rows  = (const int*)d_in[1];
  const int* cols  = (const int*)d_in[2];
  const int* seeds = (const int*)d_in[3];
  float* out = (float*)d_out;

  char* ws = (char*)d_ws;
  size_t off = 0;
  auto carve = [&](size_t bytes) -> void* {
    void* p = (void*)(ws + off);
    off += bytes;
    off = (off + 255u) & ~(size_t)255u;
    return p;
  };
  uint32_t* frontier = (uint32_t*)carve((size_t)NTOT * 4);
  uint32_t* hits     = (uint32_t*)carve((size_t)NTOT * 4);
  uint32_t* masked   = (uint32_t*)carve((size_t)NTOT_PAD * 4);
  float*    deg      = (float*)   carve((size_t)NTOT * 4);
  float*    dnorm    = (float*)   carve((size_t)NTOT * 4);
  int*      order    = (int*)     carve((size_t)NTOT * 4);
  uint8_t*  alive    = (uint8_t*) carve((size_t)EEDG);
  uint32_t* blkcnt   = (uint32_t*)carve((size_t)8192 * 4);
  uint32_t* scal     = (uint32_t*)carve((size_t)64);
  unsigned long long* keys = (unsigned long long*)carve((size_t)SORTN * 8);

  const int NB_PAD  = (NTOT_PAD + 255) / 256;
  const int NB_NODE = (NTOT + 255) / 256;
  const int NB_EDGE = (EEDG + 255) / 256;
  const int NB1 = NTOT_PAD / 4096;          // 37
  const int NB2 = (int)(SORTN / 4096);      // 4096

  // 0) init
  k_init_nodes<<<NB_PAD, 256, 0, stream>>>(frontier, hits, masked, deg);
  k_init_alive<<<NB_EDGE, 256, 0, stream>>>(alive);
  k_seed<<<(NSEEDS + 255) / 256, 256, 0, stream>>>(seeds, frontier, masked);

  // 1) BFS-style edge masking, MASK_DEPTH = 2 (TDM-streamed edge passes)
  k_edge_pass<<<1024, 32, 0, stream>>>(rows, cols, frontier, hits, alive, 1);
  k_frontier_update<<<NB_NODE, 256, 0, stream>>>(frontier, hits, masked);
  k_edge_pass<<<1024, 32, 0, stream>>>(rows, cols, frontier, hits, alive, 0);

  // 2) random augmentation of masked set
  k_sample<<<(SAMPN + 255) / 256, 256, 0, stream>>>(masked);

  // 3) tem_num + stable partition (order) via WMMA count + wave scan
  k_mask_count<<<NB1, 32, 0, stream>>>(masked, blkcnt);
  k_scan_wave<<<1, 32, 0, stream>>>(blkcnt, NB1, &scal[0]);   // scal[0] = tem_num
  k_order_write<<<NB1, 32, 0, stream>>>(masked, blkcnt, scal, order, NTOT);

  // 4) encoder: degree, norm, values; write rows/cols/enc_vals to d_out
  k_deg<<<NB_EDGE, 256, 0, stream>>>(rows, alive, deg);
  k_dnorm<<<NB_NODE, 256, 0, stream>>>(deg, dnorm);
  k_enc_out<<<NB_EDGE / 4, 256, 0, stream>>>(rows, cols, alive, dnorm, out);

  // 5) decoder: zero pad, generate hashes, bitonic sort, unique-compact
  k_dec_zero<<<(int)((3L * HLEN / 4 + 255) / 256), 256, 0, stream>>>(out);
  k_hashgen<<<(int)(SORTN / 256), 256, 0, stream>>>(rows, cols, alive, order,
                                                    scal, keys);
  // fused LDS stages k=2..4096 (78 passes in one launch)
  k_bitonic_presort<<<(int)(SORTN / 4096), 512, 0, stream>>>(keys);
  // remaining stages: global passes only for j>=4096, LDS finisher for j<=2048
  for (unsigned k = 8192; k <= SORTN; k <<= 1) {
    for (unsigned j = k >> 1; j >= 4096; j >>= 1)
      k_bitonic<<<(int)(SORTN / 256), 256, 0, stream>>>(keys, j, k);
    k_bitonic_lds<<<(int)(SORTN / 4096), 512, 0, stream>>>(keys, k);
  }

  k_uniq_count<<<NB2, 32, 0, stream>>>(keys, blkcnt);
  k_scan_wave<<<1, 32, 0, stream>>>(blkcnt, NB2, &scal[1]);
  k_uniq_write<<<NB2, 32, 0, stream>>>(keys, blkcnt, out);
}
